// MultiHeadLinearAttention_23880018166317
// MI455X (gfx1250) — compile-verified, matrix operand order checked by reading, not run on hardware
//
#include <hip/hip_runtime.h>
#include <hip/hip_bf16.h>

// ---------------- types ----------------
typedef __attribute__((ext_vector_type(16))) __bf16 v16bf;
typedef __attribute__((ext_vector_type(8)))  float  v8f;

union Frag { uint4 u[2]; v16bf v; };
union Pack4 { __bf16 h[4]; uint2 u; };

// Problem constants
#define Bb_   4
#define T_    4096
#define C_    1024
#define H_    16
#define HS_   64
#define K_    256
#define BT_   (Bb_ * T_)   // 16384

// ---------------- fp32 -> bf16 elementwise (n % 4 == 0) ----------------
__global__ __launch_bounds__(256) void cvt_f32_bf16(const float* __restrict__ in,
                                                    __bf16* __restrict__ out, long n) {
  long i = ((long)blockIdx.x * blockDim.x + threadIdx.x) * 4;
  if (i + 3 < n) {
    float4 v = *(const float4*)(in + i);
    Pack4 p;
    p.h[0] = (__bf16)v.x; p.h[1] = (__bf16)v.y;
    p.h[2] = (__bf16)v.z; p.h[3] = (__bf16)v.w;
    *(uint2*)(out + i) = p.u;
  }
}

// ---------------- W[h][c][s] fp32 -> Wt[h][s][c] bf16 ----------------
__global__ __launch_bounds__(256) void cvt_wT(const float* __restrict__ in,
                                              __bf16* __restrict__ out, long n) {
  long i = (long)blockIdx.x * blockDim.x + threadIdx.x;
  if (i < n) {
    int c = (int)(i % C_);
    long t = i / C_;
    int s = (int)(t % HS_);
    int h = (int)(t / HS_);
    out[i] = (__bf16)in[((long)h * C_ + c) * HS_ + s];
  }
}

// ------- E[hb][t][r] fp32 -> Et[hb][r][t] bf16 (32x32 LDS tile transpose) -------
__global__ __launch_bounds__(256) void cvt_ET(const float* __restrict__ E,
                                              __bf16* __restrict__ Et) {
  __shared__ float tile[32][33];
  int hb = blockIdx.z;
  int tb = blockIdx.x * 32;   // t base
  int rb = blockIdx.y * 32;   // r base
  for (int i = 0; i < 32; i += 8) {
    int t = tb + threadIdx.y + i;
    int r = rb + threadIdx.x;
    tile[threadIdx.y + i][threadIdx.x] = E[((long)hb * T_ + t) * K_ + r];
  }
  __syncthreads();
  for (int i = 0; i < 32; i += 8) {
    int r = rb + threadIdx.y + i;
    int t = tb + threadIdx.x;
    Et[((long)hb * K_ + r) * T_ + t] = (__bf16)tile[threadIdx.x][threadIdx.y + i];
  }
}

// ---------------- generic batched bf16 WMMA GEMM ----------------
// C[m][n] = sum_k A[m][k] * B[n][k]   (both operands K-major row layouts)
// Block: 128 threads (4 waves). Tile: (64*MT) x 64. Each wave: MT A-frags x 4 B-frags,
// double-buffered LDS, one barrier per 32-K chunk.
// flags: 1 = store transposed (C[n*ldc+m]), 2 = fp32 output (+bias)
template <int MT>
__global__ __launch_bounds__(128)
void gemm_bf16(const __bf16* __restrict__ A, long lda, long sAh, long sAb,
               const __bf16* __restrict__ Bm, long ldb, long sBh, long sBb,
               void* __restrict__ Cp, long ldc, long sCh, long sCb,
               int Kd, int nbInner, int flags, const float* __restrict__ bias) {
  __shared__ __bf16 Al[2][64 * MT][40];
  __shared__ __bf16 Bl[2][64][40];
  int batch = blockIdx.z;
  int h = batch / nbInner, b = batch % nbInner;
  const __bf16* Ab = A + (long)h * sAh + (long)b * sAb + (long)blockIdx.x * (64 * MT) * lda;
  const __bf16* Bp = Bm + (long)h * sBh + (long)b * sBb + (long)blockIdx.y * 64 * ldb;
  int tid = threadIdx.x;
  int wave = tid >> 5, lane = tid & 31;
  int lcol = lane & 15, hi = lane >> 4;

  int lr = tid >> 1;            // 0..63 (row for cooperative loads)
  int lc0 = (tid & 1) * 16;     // 0 or 16
  const int nk = Kd >> 5;

  uint4 ra0[MT], ra1[MT], rb0, rb1;

  auto gload = [&](int kc) {
#pragma unroll
    for (int m = 0; m < MT; ++m) {
      const __bf16* sa = Ab + (long)(m * 64 + lr) * lda + kc + lc0;
      ra0[m] = *(const uint4*)sa;
      ra1[m] = *(const uint4*)(sa + 8);
    }
    const __bf16* sb = Bp + (long)lr * ldb + kc + lc0;
    rb0 = *(const uint4*)sb;
    rb1 = *(const uint4*)(sb + 8);
  };
  auto sstore = [&](int buf) {
#pragma unroll
    for (int m = 0; m < MT; ++m) {
      *(uint4*)&Al[buf][m * 64 + lr][lc0]     = ra0[m];
      *(uint4*)&Al[buf][m * 64 + lr][lc0 + 8] = ra1[m];
    }
    *(uint4*)&Bl[buf][lr][lc0]     = rb0;
    *(uint4*)&Bl[buf][lr][lc0 + 8] = rb1;
  };

  v8f acc[MT][4];
#pragma unroll
  for (int m = 0; m < MT; ++m)
#pragma unroll
    for (int nt = 0; nt < 4; ++nt) acc[m][nt] = (v8f){};

  gload(0);
  sstore(0);
  __syncthreads();

  for (int it = 0; it < nk; ++it) {
    int cur = it & 1;
    if (it + 1 < nk) gload((it + 1) * 32);

    Frag fa[MT];
#pragma unroll
    for (int m = 0; m < MT; ++m) {
      int ar = m * 64 + wave * 16 + lcol;
      fa[m].u[0] = *(const uint4*)&Al[cur][ar][hi * 8];
      fa[m].u[1] = *(const uint4*)&Al[cur][ar][hi * 8 + 16];
    }
#pragma unroll
    for (int nt = 0; nt < 4; ++nt) {
      Frag fb;
      int br = nt * 16 + lcol;
      fb.u[0] = *(const uint4*)&Bl[cur][br][hi * 8];
      fb.u[1] = *(const uint4*)&Bl[cur][br][hi * 8 + 16];
#pragma unroll
      for (int m = 0; m < MT; ++m)
        acc[m][nt] = __builtin_amdgcn_wmma_f32_16x16x32_bf16(
            false, fa[m].v, false, fb.v, (short)0, acc[m][nt], false, false);
    }
    if (it + 1 < nk) {
      sstore(cur ^ 1);
      __syncthreads();
    }
  }

  long cbase = (long)h * sCh + (long)b * sCb;
  int colBase = blockIdx.y * 64;
#pragma unroll
  for (int m = 0; m < MT; ++m) {
    int rowBase = blockIdx.x * (64 * MT) + m * 64 + wave * 16 + hi * 8;
    if (flags & 2) {
      float* Cf = (float*)Cp;
#pragma unroll
      for (int nt = 0; nt < 4; ++nt) {
        int col = colBase + nt * 16 + lcol;
        float bv = bias ? bias[col] : 0.f;
#pragma unroll
        for (int j = 0; j < 8; ++j)
          Cf[cbase + (long)(rowBase + j) * ldc + col] = acc[m][nt][j] + bv;
      }
    } else {
      __bf16* Cb = (__bf16*)Cp;
#pragma unroll
      for (int nt = 0; nt < 4; ++nt) {
        int col = colBase + nt * 16 + lcol;
#pragma unroll
        for (int j = 0; j < 8; ++j) {
          int row = rowBase + j;
          long idx = (flags & 1) ? ((long)col * ldc + row) : ((long)row * ldc + col);
          Cb[cbase + idx] = (__bf16)acc[m][nt][j];
        }
      }
    }
  }
}

// ---------------- fused attention: S = q*kp^T, mask, softmax, O = P*vp ----------------
// grid: (T/64, H*B), block 128 (4 waves); per block handles 64 query rows of one (h,b)
__global__ __launch_bounds__(128)
void attn_fused(const __bf16* __restrict__ q, const __bf16* __restrict__ kp,
                const __bf16* __restrict__ vpT, __bf16* __restrict__ attn) {
  __shared__ __bf16 kp_s[K_][72];     // [256][64] padded
  __shared__ __bf16 vp_s[HS_][264];   // [64][256] padded
  __shared__ __bf16 P_s[64][264];     // softmaxed probs, bf16
  int hb = blockIdx.y;
  int h = hb >> 2, b = hb & 3;        // B_ == 4
  int t0 = blockIdx.x * 64;
  int tid = threadIdx.x, wave = tid >> 5, lane = tid & 31;
  int lcol = lane & 15, hi = lane >> 4;

  // stage kp [256][64] and vpT [64][256] into LDS (b128 loads)
  const __bf16* kpb = kp + (long)hb * K_ * HS_;
  const __bf16* vpb = vpT + (long)hb * HS_ * K_;
  for (int i = tid; i < 2048; i += 128) {
    int r = i >> 3, c = (i & 7) * 8;
    *(uint4*)&kp_s[r][c] = *(const uint4*)(kpb + (long)r * HS_ + c);
  }
  for (int i = tid; i < 2048; i += 128) {
    int r = i >> 5, c = (i & 31) * 8;
    *(uint4*)&vp_s[r][c] = *(const uint4*)(vpb + (long)r * K_ + c);
  }

  // q fragments for this wave's 16 rows (K-dim = HS = 64 -> 2 chunks)
  const __bf16* qb = q + (long)h * BT_ * HS_ + ((long)b * T_ + t0) * HS_;
  int qrow = wave * 16 + lcol;
  Frag fq[2];
#pragma unroll
  for (int c = 0; c < 2; ++c) {
    const __bf16* p = qb + (long)qrow * HS_ + c * 32 + hi * 8;
    fq[c].u[0] = *(const uint4*)p;
    fq[c].u[1] = *(const uint4*)(p + 16);
  }
  __syncthreads();

  // S[16 rows][256] in 16 accumulators
  v8f S[16];
#pragma unroll
  for (int rt = 0; rt < 16; ++rt) S[rt] = (v8f){};
#pragma unroll
  for (int c = 0; c < 2; ++c) {
#pragma unroll
    for (int rt = 0; rt < 16; ++rt) {
      Frag fb;
      int br = rt * 16 + lcol;
      fb.u[0] = *(const uint4*)&kp_s[br][c * 32 + hi * 8];
      fb.u[1] = *(const uint4*)&kp_s[br][c * 32 + hi * 8 + 16];
      S[rt] = __builtin_amdgcn_wmma_f32_16x16x32_bf16(
          false, fq[c].v, false, fb.v, (short)0, S[rt], false, false);
    }
  }

  // masked softmax over r (per row), scale 1/sqrt(K)=0.0625
  const float scale = 0.0625f;
#pragma unroll
  for (int j = 0; j < 8; ++j) {
    int trow = t0 + wave * 16 + hi * 8 + j;
    float mx = -3.4e38f;
#pragma unroll
    for (int rt = 0; rt < 16; ++rt) {
      int r = rt * 16 + lcol;
      float val = (r <= trow) ? S[rt][j] * scale : -3.4e38f;
      S[rt][j] = val;
      mx = fmaxf(mx, val);
    }
#pragma unroll
    for (int m = 8; m >= 1; m >>= 1) mx = fmaxf(mx, __shfl_xor(mx, m, 32));
    float sum = 0.f;
#pragma unroll
    for (int rt = 0; rt < 16; ++rt) {
      float e = __expf(S[rt][j] - mx);
      S[rt][j] = e;
      sum += e;
    }
#pragma unroll
    for (int m = 8; m >= 1; m >>= 1) sum += __shfl_xor(sum, m, 32);
    float rinv = 1.f / sum;
#pragma unroll
    for (int rt = 0; rt < 16; ++rt)
      P_s[wave * 16 + hi * 8 + j][rt * 16 + lcol] = (__bf16)(S[rt][j] * rinv);
  }
  __syncthreads();

  // O[16][64] = P[16][256] x vp   (K-dim 256 -> 8 chunks)
  v8f O[4] = {};
  int prow = wave * 16 + lcol;
  for (int kc = 0; kc < K_; kc += 32) {
    Frag fp;
    fp.u[0] = *(const uint4*)&P_s[prow][kc + hi * 8];
    fp.u[1] = *(const uint4*)&P_s[prow][kc + hi * 8 + 16];
#pragma unroll
    for (int nt = 0; nt < 4; ++nt) {
      Frag fv;
      int vr = nt * 16 + lcol;
      fv.u[0] = *(const uint4*)&vp_s[vr][kc + hi * 8];
      fv.u[1] = *(const uint4*)&vp_s[vr][kc + hi * 8 + 16];
      O[nt] = __builtin_amdgcn_wmma_f32_16x16x32_bf16(
          false, fp.v, false, fv.v, (short)0, O[nt], false, false);
    }
  }

  // store to attn[b*T + t][h*64 + s] bf16
  __bf16* ob = attn + ((long)b * T_ + t0) * C_ + h * HS_;
#pragma unroll
  for (int nt = 0; nt < 4; ++nt) {
    int col = nt * 16 + lcol;
#pragma unroll
    for (int j = 0; j < 8; ++j)
      ob[(long)(wave * 16 + hi * 8 + j) * C_ + col] = (__bf16)O[nt][j];
  }
}

// ---------------- host side ----------------
extern "C" void kernel_launch(void* const* d_in, const int* in_sizes, int n_in,
                              void* d_out, int out_size, void* d_ws, size_t ws_size,
                              hipStream_t stream) {
  const float* x  = (const float*)d_in[0];   // [B,T,C]
  const float* WQ = (const float*)d_in[1];   // [H,C,HS]
  const float* WK = (const float*)d_in[2];
  const float* WV = (const float*)d_in[3];
  const float* E  = (const float*)d_in[4];   // [H,B,T,K]
  const float* Wp = (const float*)d_in[5];   // [C,C]
  const float* bp = (const float*)d_in[6];   // [C]
  float* out = (float*)d_out;                // [B,T,C]

  char* w = (char*)d_ws;
  auto alloc = [&](size_t bytes) {
    char* p = w;
    w += (bytes + 255) & ~(size_t)255;
    return p;
  };
  __bf16* xbf   = (__bf16*)alloc((size_t)BT_ * C_ * 2);        // 32 MB
  __bf16* WtQ   = (__bf16*)alloc((size_t)H_ * HS_ * C_ * 2);
  __bf16* WtK   = (__bf16*)alloc((size_t)H_ * HS_ * C_ * 2);
  __bf16* WtV   = (__bf16*)alloc((size_t)H_ * HS_ * C_ * 2);
  __bf16* Wpbf  = (__bf16*)alloc((size_t)C_ * C_ * 2);
  __bf16* Et    = (__bf16*)alloc((size_t)H_ * Bb_ * K_ * T_ * 2);  // 128 MB
  __bf16* qb    = (__bf16*)alloc((size_t)H_ * BT_ * HS_ * 2);
  __bf16* kT    = (__bf16*)alloc((size_t)H_ * HS_ * BT_ * 2);
  __bf16* vT    = (__bf16*)alloc((size_t)H_ * HS_ * BT_ * 2);
  __bf16* kpb   = (__bf16*)alloc((size_t)H_ * Bb_ * K_ * HS_ * 2);
  __bf16* vpTb  = (__bf16*)alloc((size_t)H_ * Bb_ * HS_ * K_ * 2);
  __bf16* attnb = (__bf16*)alloc((size_t)BT_ * C_ * 2);
  (void)ws_size; (void)n_in; (void)in_sizes; (void)out_size;

  // 1) conversions
  {
    long n = (long)BT_ * C_;
    cvt_f32_bf16<<<dim3((unsigned)(n / 4 / 256)), dim3(256), 0, stream>>>(x, xbf, n);
  }
  {
    long n = (long)C_ * C_;
    cvt_f32_bf16<<<dim3((unsigned)(n / 4 / 256)), dim3(256), 0, stream>>>(Wp, Wpbf, n);
  }
  {
    long n = (long)H_ * HS_ * C_;
    unsigned g = (unsigned)((n + 255) / 256);
    cvt_wT<<<dim3(g), dim3(256), 0, stream>>>(WQ, WtQ, n);
    cvt_wT<<<dim3(g), dim3(256), 0, stream>>>(WK, WtK, n);
    cvt_wT<<<dim3(g), dim3(256), 0, stream>>>(WV, WtV, n);
  }
  cvt_ET<<<dim3(T_ / 32, K_ / 32, H_ * Bb_), dim3(32, 8), 0, stream>>>(E, Et);

  // 2) QKV projections: [BT x C] x [HS x C]^row -> per-head outputs
  // q: normal [h][bt][s];  k,v: transposed [h][s][bt]
  gemm_bf16<2><<<dim3(BT_ / 128, HS_ / 64, H_), dim3(128), 0, stream>>>(
      xbf, C_, 0, 0, WtQ, C_, (long)HS_ * C_, 0,
      qb, HS_, (long)BT_ * HS_, 0, C_, 1, 0, nullptr);
  gemm_bf16<2><<<dim3(BT_ / 128, HS_ / 64, H_), dim3(128), 0, stream>>>(
      xbf, C_, 0, 0, WtK, C_, (long)HS_ * C_, 0,
      kT, BT_, (long)HS_ * BT_, 0, C_, 1, 1, nullptr);
  gemm_bf16<2><<<dim3(BT_ / 128, HS_ / 64, H_), dim3(128), 0, stream>>>(
      xbf, C_, 0, 0, WtV, C_, (long)HS_ * C_, 0,
      vT, BT_, (long)HS_ * BT_, 0, C_, 1, 1, nullptr);

  // 3) kp[h][b][r][s] = Et[h][b][r][:] . kT[h][s][b*T + :]   (Kd = T)
  gemm_bf16<2><<<dim3(K_ / 128, HS_ / 64, H_ * Bb_), dim3(128), 0, stream>>>(
      Et, T_, (long)Bb_ * K_ * T_, (long)K_ * T_,
      kT, BT_, (long)HS_ * BT_, T_,
      kpb, HS_, (long)Bb_ * K_ * HS_, (long)K_ * HS_, T_, Bb_, 0, nullptr);
  //    vpT[h][b][s][r] = vT[h][s][b*T + :] . Et[h][b][r][:]   (M = 64 -> MT=1)
  gemm_bf16<1><<<dim3(HS_ / 64, K_ / 64, H_ * Bb_), dim3(128), 0, stream>>>(
      vT, BT_, (long)HS_ * BT_, T_,
      Et, T_, (long)Bb_ * K_ * T_, (long)K_ * T_,
      vpTb, K_, (long)Bb_ * HS_ * K_, (long)HS_ * K_, T_, Bb_, 0, nullptr);

  // 4) fused masked-softmax attention, writes attnb[bt][h*HS + s]
  attn_fused<<<dim3(T_ / 64, H_ * Bb_), dim3(128), 0, stream>>>(qb, kpb, vpTb, attnb);

  // 5) output projection: out[bt][i] = attn[bt][:] . Wp[i][:] + bp[i]  (fp32 out)
  gemm_bf16<2><<<dim3(BT_ / 128, C_ / 64, 1), dim3(128), 0, stream>>>(
      attnb, C_, 0, 0, Wpbf, C_, 0, 0,
      out, C_, 0, 0, C_, 1, 2, bp);
}
